// QLSTM__gen558_65481071399130
// MI455X (gfx1250) — compile-verified, matrix operand order checked by reading, not run on hardware
//
#include <hip/hip_runtime.h>
#include <hip/hip_bf16.h>

// ---------------------------------------------------------------------------
// QLSTM for MI455X (gfx1250).
//   T=256, B=128, D_IN=512, H=256, C=768, 4 gates -> N=1024
// Phase 1: Zx = x @ Wx^T + (b + theta)   (bf16 WMMA GEMM, ping-pong
//          software pipeline, tile-major output for coalesced store/reload)
// Phase 2: per-batch-group-of-16 persistent recurrence (8 independent WGs,
//          no grid sync; h/c state + gate buffers in LDS; WMMA for h @ Wh^T,
//          pipelined fragment loads, wave32 shfl cumprod scans)
// ---------------------------------------------------------------------------

#define QT   256
#define QB   128
#define QDIN 512
#define QH   256
#define QN   1024   // 4 * H

typedef __attribute__((ext_vector_type(16))) __bf16 v16bf;
typedef __attribute__((ext_vector_type(8)))  float  v8f;

union ABFrag { v16bf v; unsigned u[8]; };

__device__ __forceinline__ unsigned short f2bf(float f) {
  unsigned u = __float_as_uint(f);
  u += 0x7FFFu + ((u >> 16) & 1u);   // round-to-nearest-even
  return (unsigned short)(u >> 16);
}

// K index pattern for 16-bit A/B fragments (ISA 7.12.2, 16x32 bf16):
// lane<16: VGPR0-3 -> K=0..7, VGPR4-7 -> K=16..23 ; lane>=16: +8
// Per lane this is two contiguous 16B runs -> compiler merges to b128 loads.
__device__ __forceinline__ int kpat(int v, int khalf) {
  return (v < 4) ? (v * 2 + khalf * 8) : (16 + (v - 4) * 2 + khalf * 8);
}

// ---------------------------------------------------------------------------
// Pack weights to bf16 (Wx: [1024][512], Wh: [1024][256]) and fuse bias+theta.
// ---------------------------------------------------------------------------
__global__ void k_pack(const float* __restrict__ Wf, const float* __restrict__ bf_, const float* __restrict__ tf_,
                       const float* __restrict__ Wi, const float* __restrict__ bi_, const float* __restrict__ ti_,
                       const float* __restrict__ Wg, const float* __restrict__ bg_, const float* __restrict__ tg_,
                       const float* __restrict__ Wo, const float* __restrict__ bo_, const float* __restrict__ to_,
                       unsigned short* __restrict__ wxbf, unsigned short* __restrict__ whbf,
                       float* __restrict__ biasv)
{
  int idx = blockIdx.x * blockDim.x + threadIdx.x;
  if (idx >= QN * (QDIN + QH)) return;
  int n = idx / (QDIN + QH);
  int c = idx % (QDIN + QH);
  int g = n >> 8, h = n & 255;
  const float* W = (g == 0) ? Wf : (g == 1) ? Wi : (g == 2) ? Wg : Wo;
  float w = W[h * (QDIN + QH) + c];
  if (c < QDIN) wxbf[n * QDIN + c] = f2bf(w);
  else          whbf[n * QH + (c - QDIN)] = f2bf(w);
  if (c == 0) {
    const float* bb = (g == 0) ? bf_ : (g == 1) ? bi_ : (g == 2) ? bg_ : bo_;
    const float* tt = (g == 0) ? tf_ : (g == 1) ? ti_ : (g == 2) ? tg_ : to_;
    biasv[n] = bb[h] + tt[h];
  }
}

// ---------------------------------------------------------------------------
// Convert inputs x [T*B, 512] f32 -> bf16 (packed pairs, 4 elems/thread).
// ---------------------------------------------------------------------------
__global__ void k_cvtx(const float* __restrict__ x, unsigned short* __restrict__ xbf)
{
  int i = blockIdx.x * blockDim.x + threadIdx.x;
  int base = i * 4;                       // total = 32768*512, divisible by 4
  float a = x[base], b = x[base + 1], c = x[base + 2], d = x[base + 3];
  unsigned* o = (unsigned*)xbf;
  o[(base >> 1)]     = (unsigned)f2bf(a) | ((unsigned)f2bf(b) << 16);
  o[(base >> 1) + 1] = (unsigned)f2bf(c) | ((unsigned)f2bf(d) << 16);
}

// ---------------------------------------------------------------------------
// Phase 1 GEMM: Zx = xbf @ Wx^T + bias, stored TILE-MAJOR:
//   zx[((mtile*64 + ntile) * 256) + r*32 + lane]  (WMMA C-fragment order)
// Wave computes one 16(M) x 128(N) strip: 8 accumulators share the A fragment.
// Ping-pong double buffers (constant indices under unroll) keep loads one
// step ahead of the consuming WMMA with zero register-copy / hazard-NOP cost.
// 2048 M-tiles x 8 N-groups = 16384 waves = 2048 blocks x 8 waves.
// ---------------------------------------------------------------------------
__global__ __launch_bounds__(256) void k_gemm_x(const unsigned short* __restrict__ xbf,
                                                const unsigned short* __restrict__ wxbf,
                                                const float* __restrict__ biasv,
                                                float* __restrict__ zx)
{
  int wave  = blockIdx.x * (blockDim.x >> 5) + (threadIdx.x >> 5);
  int lane  = threadIdx.x & 31;
  int mt    = wave >> 3;            // 0..2047
  int ng    = wave & 7;             // 0..7  (128-col group)
  int m0    = mt * 16;
  int n0    = ng * 128;
  int mrow  = lane & 15;
  int khalf = lane >> 4;

  const unsigned* xrow  = (const unsigned*)xbf + (size_t)(m0 + mrow) * (QDIN / 2);
  const unsigned* wbase = (const unsigned*)wxbf;

  v8f acc[8];
#pragma unroll
  for (int j = 0; j < 8; ++j) acc[j] = v8f{};

  auto loadA = [&](ABFrag& f, int kk) {
#pragma unroll
    for (int v = 0; v < 8; ++v) f.u[v] = xrow[(kk + kpat(v, khalf)) >> 1];
  };
  auto loadB = [&](ABFrag& f, int kk, int j) {
    const unsigned* wrow = wbase + (size_t)(n0 + j * 16 + (lane & 15)) * (QDIN / 2);
#pragma unroll
    for (int v = 0; v < 8; ++v) f.u[v] = wrow[(kk + kpat(v, khalf)) >> 1];
  };

  ABFrag aa[2], bb[2];
  aa[0] = {}; aa[1] = {}; bb[0] = {}; bb[1] = {};
  loadA(aa[0], 0);
  loadB(bb[0], 0, 0);
  for (int kk2 = 0; kk2 < QDIN; kk2 += 64) {
#pragma unroll
    for (int u = 0; u < 2; ++u) {
      int kk = kk2 + u * 32;
#pragma unroll
      for (int j = 0; j < 8; ++j) {
        if (j < 7) {
          loadB(bb[(j + 1) & 1], kk, j + 1);        // while WMMA j runs
        } else if (kk + 32 < QDIN) {
          loadB(bb[0], kk + 32, 0);                 // wraps to next K-step
          loadA(aa[u ^ 1], kk + 32);
        }
        acc[j] = __builtin_amdgcn_wmma_f32_16x16x32_bf16(
            false, aa[u].v, false, bb[j & 1].v, (short)0, acc[j], false, false);
      }
    }
  }

#pragma unroll
  for (int j = 0; j < 8; ++j) {
    int n = n0 + j * 16 + (lane & 15);
    float bias = biasv[n];
    float* tile = zx + ((size_t)mt * 64 + (size_t)ng * 8 + j) * 256;
#pragma unroll
    for (int r = 0; r < 8; ++r)
      tile[r * 32 + lane] = acc[j][r] + bias;   // coalesced 128B/wave stores
  }
}

// ---------------------------------------------------------------------------
// Phase 2: persistent recurrence. 8 blocks x 1024 threads (32 waves).
// Block owns batches [16*blk, 16*blk+16); all state in LDS; T-loop with only
// workgroup barriers. Per step: 64 N-tiles (2/wave) of h@Wh^T via WMMA
// (ping-pong LDS-A / global-B fragment loads, Zx C-addend preloaded),
// cos -> LDS, wave-scan cumprod, fused LSTM update, repack h to bf16 frags.
// ---------------------------------------------------------------------------
__global__ __launch_bounds__(1024) void k_recur(const unsigned short* __restrict__ whbf,
                                                const float* __restrict__ zx,
                                                float* __restrict__ out)
{
  __shared__ unsigned hbf[16][129];        // h as packed bf16 pairs (padded)
  __shared__ float    cst[16][QH];         // cell state
  __shared__ float    gbuf[4][16][257];    // cos(mapped) -> cumprod (padded)

  int tid   = threadIdx.x;
  int lane  = tid & 31;
  int wv    = tid >> 5;                    // 0..31
  int blk   = blockIdx.x;
  int b0    = blk * 16;
  int mrow  = lane & 15;
  int khalf = lane >> 4;

  for (int i = tid; i < 16 * 129; i += 1024) ((unsigned*)hbf)[i] = 0u;
  for (int i = tid; i < 16 * QH;  i += 1024) ((float*)cst)[i] = 0.f;
  __syncthreads();

  for (int t = 0; t < QT; ++t) {
    // Tile-major Zx strip for this (t, batch-group): 64 tiles x 1KB, contiguous.
    const float* zxt = zx + ((size_t)(t * 8 + blk) * 64) * 256;

    // Prefetch next step's contiguous 64KB strip (gfx1250 global_prefetch).
    if (t + 1 < QT) {
      const float* nxt = zx + ((size_t)((t + 1) * 8 + blk) * 64) * 256 + (size_t)tid * 16;
      __builtin_prefetch(nxt, 0, 1);
    }

    // ---- GEMM: mapped = Zx[t] + h @ Wh^T ; gbuf = cos(mapped) ----
#pragma unroll
    for (int half = 0; half < 2; ++half) {
      int nt = wv + half * 32;             // 0..63
      int g  = nt >> 4;
      int h0 = (nt & 15) * 16;
      int n  = g * QH + h0 + (lane & 15);
      const unsigned* wrow = (const unsigned*)whbf + (size_t)n * (QH / 2);
      const float*    ztile = zxt + (size_t)nt * 256;

      // C-addend: coalesced, issued early so it overlaps the whole K-loop.
      float zc[8];
#pragma unroll
      for (int r = 0; r < 8; ++r) zc[r] = ztile[r * 32 + lane];

      auto loadA = [&](ABFrag& f, int kk) {
#pragma unroll
        for (int v = 0; v < 8; ++v) f.u[v] = hbf[mrow][(kk + kpat(v, khalf)) >> 1];
      };
      auto loadB = [&](ABFrag& f, int kk) {
#pragma unroll
        for (int v = 0; v < 8; ++v) f.u[v] = wrow[(kk + kpat(v, khalf)) >> 1];
      };

      v8f acc = {};
      ABFrag aa[2], bb[2];
      aa[0] = {}; aa[1] = {}; bb[0] = {}; bb[1] = {};
      loadA(aa[0], 0);
      loadB(bb[0], 0);
#pragma unroll
      for (int ki = 0; ki < 8; ++ki) {
        if (ki < 7) {
          loadA(aa[(ki + 1) & 1], (ki + 1) * 32);
          loadB(bb[(ki + 1) & 1], (ki + 1) * 32);
        }
        acc = __builtin_amdgcn_wmma_f32_16x16x32_bf16(
            false, aa[ki & 1].v, false, bb[ki & 1].v, (short)0, acc, false, false);
      }
#pragma unroll
      for (int r = 0; r < 8; ++r) {
        int m = r + 8 * (lane >> 4);
        gbuf[g][m][h0 + (lane & 15)] = __cosf(acc[r] + zc[r]);
      }
    }
    __syncthreads();

    // ---- cumprod along H: 64 rows (gate,b), 2 per wave, wave32 scan ----
#pragma unroll
    for (int rr = 0; rr < 2; ++rr) {
      int row = wv * 2 + rr;
      int g = row >> 4, b = row & 15;
      float carry = 1.f;
      for (int ch = 0; ch < 8; ++ch) {
        float xv = gbuf[g][b][ch * 32 + lane];
#pragma unroll
        for (int off = 1; off < 32; off <<= 1) {
          float y = __shfl_up(xv, off, 32);
          if (lane >= off) xv *= y;
        }
        xv *= carry;
        carry = __shfl(xv, 31, 32);
        gbuf[g][b][ch * 32 + lane] = xv;
      }
    }
    __syncthreads();

    // ---- fused LSTM update: 16x256 elems, 4 per thread; repack h->bf16 ----
    {
      int b  = tid >> 6;
      int hh = (tid & 63) * 4;
      unsigned packed[2];
#pragma unroll
      for (int q = 0; q < 4; ++q) {
        int h = hh + q;
        float F = gbuf[0][b][h], I = gbuf[1][b][h];
        float G = gbuf[2][b][h], O = gbuf[3][b][h];
        float fg = 1.f / (1.f + __expf(-F));
        float ig = 1.f / (1.f + __expf(-I));
        float gg = tanhf(G);
        float og = 1.f / (1.f + __expf(-O));
        float c  = fg * cst[b][h] + ig * gg;
        cst[b][h] = c;
        float hv = og * tanhf(c);
        out[((size_t)t * QB + (b0 + b)) * QH + h] = hv;
        if (t == QT - 1) {
          size_t tail = (size_t)QT * QB * QH;
          out[tail + (size_t)(b0 + b) * QH + h] = hv;                  // hx
          out[tail + (size_t)QB * QH + (size_t)(b0 + b) * QH + h] = c; // cx
        }
        unsigned short hb = f2bf(hv);
        if (q & 1) packed[q >> 1] |= ((unsigned)hb << 16);
        else       packed[q >> 1]  = (unsigned)hb;
      }
      hbf[b][(hh >> 1)]     = packed[0];
      hbf[b][(hh >> 1) + 1] = packed[1];
    }
    __syncthreads();
  }
}

// ---------------------------------------------------------------------------
extern "C" void kernel_launch(void* const* d_in, const int* in_sizes, int n_in,
                              void* d_out, int out_size, void* d_ws, size_t ws_size,
                              hipStream_t stream)
{
  (void)in_sizes; (void)n_in; (void)out_size; (void)ws_size;
  const float* x   = (const float*)d_in[0];
  const float* Wf  = (const float*)d_in[1];
  const float* bf_ = (const float*)d_in[2];
  const float* tf_ = (const float*)d_in[3];
  const float* Wi  = (const float*)d_in[4];
  const float* bi_ = (const float*)d_in[5];
  const float* ti_ = (const float*)d_in[6];
  const float* Wg  = (const float*)d_in[7];
  const float* bg_ = (const float*)d_in[8];
  const float* tg_ = (const float*)d_in[9];
  const float* Wo  = (const float*)d_in[10];
  const float* bo_ = (const float*)d_in[11];
  const float* to_ = (const float*)d_in[12];

  char* ws = (char*)d_ws;
  size_t off = 0;
  auto carve = [&](size_t bytes) -> void* {
    void* p = ws + off;
    off += (bytes + 255) & ~(size_t)255;
    return p;
  };
  float*          zx    = (float*)carve((size_t)QT * QB * QN * sizeof(float));     // 134 MB
  unsigned short* xbf   = (unsigned short*)carve((size_t)QT * QB * QDIN * 2);      // 33.5 MB
  unsigned short* wxbf  = (unsigned short*)carve((size_t)QN * QDIN * 2);           // 1 MB
  unsigned short* whbf  = (unsigned short*)carve((size_t)QN * QH * 2);             // 0.5 MB
  float*          biasv = (float*)carve((size_t)QN * sizeof(float));

  k_pack<<<(QN * (QDIN + QH) + 255) / 256, 256, 0, stream>>>(
      Wf, bf_, tf_, Wi, bi_, ti_, Wg, bg_, tg_, Wo, bo_, to_, wxbf, whbf, biasv);

  k_cvtx<<<((size_t)QT * QB * QDIN / 4 + 255) / 256, 256, 0, stream>>>(x, xbf);

  // 2048 M-tiles x 8 N-groups = 16384 waves = 2048 blocks x 8 waves
  k_gemm_x<<<2048, 256, 0, stream>>>(xbf, wxbf, biasv, zx);

  // 8 independent batch groups, each a persistent 1024-thread workgroup
  k_recur<<<QB / 16, 1024, 0, stream>>>(whbf, zx, (float*)d_out);
}
